// GAT_14104672600429
// MI455X (gfx1250) — compile-verified, matrix operand order checked by reading, not run on hardware
//
#include <hip/hip_runtime.h>
#include <hip/hip_bf16.h>

typedef __attribute__((ext_vector_type(8)))  float  v8f;
typedef __attribute__((ext_vector_type(16))) __bf16 v16bf;

// ---------- helpers ----------
static __device__ __forceinline__ unsigned short f2bf(float f) {
  unsigned u = __float_as_uint(f);
  u = (u + 0x7FFFu + ((u >> 16) & 1u)) >> 16;        // RNE to bf16
  return (unsigned short)u;
}
static __device__ __forceinline__ unsigned f2bf_pk(float lo, float hi) {
  return (unsigned)f2bf(lo) | ((unsigned)f2bf(hi) << 16);
}

// ---------- fp32 buffer -> bf16 buffer (n multiple of 4) ----------
__global__ void f32_to_bf16(const float* __restrict__ src, unsigned* __restrict__ dst,
                            int n4) {  // n4 = n/4, dst as packed uint pairs
  for (int i = blockIdx.x * blockDim.x + threadIdx.x; i < n4;
       i += gridDim.x * blockDim.x) {
    float4 f = reinterpret_cast<const float4*>(src)[i];
    dst[2 * i]     = f2bf_pk(f.x, f.y);
    dst[2 * i + 1] = f2bf_pk(f.z, f.w);
  }
}

// ---------- W (fp32 [K,Nout]) -> Wt (bf16, column-major [Nout,K]) ----------
__global__ void convert_wt(const float* __restrict__ W, unsigned short* __restrict__ Wt,
                           int K, int Nout) {
  int i = blockIdx.x * blockDim.x + threadIdx.x;
  if (i >= K * Nout) return;
  int n = i / K, k = i % K;
  Wt[(size_t)n * K + k] = f2bf(W[(size_t)k * Nout + n]);
}

// ---------- WMMA GEMM: C[M,NB*16] = A_bf16[M,KDIM] x Wt ----------
template <int KDIM, int NB>
__global__ void gemm_wmma(const unsigned short* __restrict__ A,
                          const unsigned short* __restrict__ Wt,
                          float* __restrict__ C, int mTiles, int ldc) {
  const int wave = threadIdx.x >> 5;
  const int lane = threadIdx.x & 31;
  const int rowTile = blockIdx.x * (blockDim.x >> 5) + wave;
  if (rowTile >= mTiles) return;            // wave-uniform: EXEC stays all-ones
  const int row0   = rowTile * 16;
  const int laneM  = lane & 15;
  const int hiHalf = lane >> 4;             // 0: K {0..7,16..23}; 1: K {8..15,24..31}
  const unsigned short* arow = A + (size_t)(row0 + laneM) * KDIM;

  v8f acc[NB] = {};

#pragma unroll
  for (int kc = 0; kc < KDIM; kc += 32) {
    const int kb = kc + hiHalf * 8;
    union { v16bf v; uint4 u4[2]; } af;
    af.u4[0] = *reinterpret_cast<const uint4*>(arow + kb);       // K = kb..kb+7
    af.u4[1] = *reinterpret_cast<const uint4*>(arow + kb + 16);  // K = kb+16..kb+23

#pragma unroll
    for (int nb = 0; nb < NB; ++nb) {
      union { v16bf v; uint4 u4[2]; } bf;
      const unsigned short* bcol = Wt + (size_t)(nb * 16 + laneM) * KDIM + kb;
      bf.u4[0] = *reinterpret_cast<const uint4*>(bcol);
      bf.u4[1] = *reinterpret_cast<const uint4*>(bcol + 16);
      acc[nb] = __builtin_amdgcn_wmma_f32_16x16x32_bf16(
          false, af.v, false, bf.v, (short)0, acc[nb], false, false);
    }
  }

  // C/D layout: lane -> N, VGPR r -> M (upper lanes +8)
#pragma unroll
  for (int nb = 0; nb < NB; ++nb)
#pragma unroll
    for (int r = 0; r < 8; ++r)
      C[(size_t)(row0 + r + hiHalf * 8) * ldc + nb * 16 + laneM] = acc[nb][r];
}

// ---------- per-node attention coefficients ----------
__global__ void node_attn(const float* __restrict__ h, const float* __restrict__ a_s,
                          const float* __restrict__ a_d, float* __restrict__ als,
                          float* __restrict__ ald, int N, int H, int D) {
  int i = blockIdx.x * blockDim.x + threadIdx.x;
  if (i >= N * H) return;
  int n = i / H, hh = i % H;
  const float* hp = h + (size_t)n * H * D + hh * D;
  float ss = 0.f, sd = 0.f;
  for (int d = 0; d < D; ++d) {
    float v = hp[d];
    ss += v * a_s[hh * D + d];
    sd += v * a_d[hh * D + d];
  }
  als[i] = ss;
  ald[i] = sd;
}

// ---------- init m = -inf, s = 0 ----------
__global__ void init_ms(float* __restrict__ m, float* __restrict__ s, int n) {
  int i = blockIdx.x * blockDim.x + threadIdx.x;
  if (i < n) { ((unsigned*)m)[i] = 0xFF800000u; s[i] = 0.f; }
}

__global__ void zero_buf(float* __restrict__ p, int n) {
  for (int i = blockIdx.x * blockDim.x + threadIdx.x; i < n;
       i += gridDim.x * blockDim.x)
    p[i] = 0.f;
}

// ---------- segment max over dst (float max via int atomics) ----------
__global__ void edge_max(const int* __restrict__ ei, const float* __restrict__ als,
                         const float* __restrict__ ald, float* __restrict__ m,
                         int E, int Etot, int H) {
  long long total = (long long)Etot * H;
  for (long long t = blockIdx.x * (long long)blockDim.x + threadIdx.x; t < total;
       t += (long long)gridDim.x * blockDim.x) {
    int e = (int)(t / H), hh = (int)(t % H);
    int src, dst;
    if (e < E) { src = ei[e]; dst = ei[E + e]; } else { src = dst = e - E; }
    float v = als[src * H + hh] + ald[dst * H + hh];
    v = v > 0.f ? v : 0.2f * v;                              // leaky_relu
    float* addr = m + (size_t)dst * H + hh;
    if (v >= 0.f) atomicMax((int*)addr, __float_as_int(v));
    else          atomicMin((unsigned*)addr, __float_as_uint(v));
  }
}

// ---------- segment sum of exp(e - m[dst]) ----------
__global__ void edge_sum(const int* __restrict__ ei, const float* __restrict__ als,
                         const float* __restrict__ ald, const float* __restrict__ m,
                         float* __restrict__ s, int E, int Etot, int H) {
  long long total = (long long)Etot * H;
  for (long long t = blockIdx.x * (long long)blockDim.x + threadIdx.x; t < total;
       t += (long long)gridDim.x * blockDim.x) {
    int e = (int)(t / H), hh = (int)(t % H);
    int src, dst;
    if (e < E) { src = ei[e]; dst = ei[E + e]; } else { src = dst = e - E; }
    int dh = dst * H + hh;
    float v = als[src * H + hh] + ald[dh];
    v = v > 0.f ? v : 0.2f * v;
    atomicAdd(&s[dh], __expf(v - m[dh]));
  }
}

// ---------- weighted message aggregation ----------
__global__ void edge_agg(const int* __restrict__ ei, const float* __restrict__ h,
                         const float* __restrict__ als, const float* __restrict__ ald,
                         const float* __restrict__ m, const float* __restrict__ s,
                         float* __restrict__ out, int E, int Etot, int H, int D) {
  const int HD = H * D;
  long long total = (long long)Etot * HD;
  for (long long t = blockIdx.x * (long long)blockDim.x + threadIdx.x; t < total;
       t += (long long)gridDim.x * blockDim.x) {
    int e = (int)(t / HD), f = (int)(t % HD), hh = f / D;
    int src, dst;
    if (e < E) { src = ei[e]; dst = ei[E + e]; } else { src = dst = e - E; }
    int dh = dst * H + hh;
    float v = als[src * H + hh] + ald[dh];
    v = v > 0.f ? v : 0.2f * v;
    float alpha = __expf(v - m[dh]) / s[dh];
    atomicAdd(&out[(size_t)dst * HD + f], h[(size_t)src * HD + f] * alpha);
  }
}

// ---------- bias + relu, emit bf16 activations for next layer's GEMM ----------
__global__ void bias_act_bf16(const float* __restrict__ acc, const float* __restrict__ b,
                              unsigned short* __restrict__ dst, int n, int HD) {
  for (int i = blockIdx.x * blockDim.x + threadIdx.x; i < n;
       i += gridDim.x * blockDim.x) {
    float v = acc[i] + b[i % HD];
    v = fmaxf(v, 0.f);
    dst[i] = f2bf(v);
  }
}

// ---------- final layer: bias only, fp32 in place ----------
__global__ void bias_final(float* __restrict__ out, const float* __restrict__ b,
                           int n, int HD) {
  for (int i = blockIdx.x * blockDim.x + threadIdx.x; i < n;
       i += gridDim.x * blockDim.x)
    out[i] = out[i] + b[i % HD];
}

extern "C" void kernel_launch(void* const* d_in, const int* in_sizes, int n_in,
                              void* d_out, int out_size, void* d_ws, size_t ws_size,
                              hipStream_t stream) {
  (void)n_in; (void)out_size; (void)ws_size;
  const float* x  = (const float*)d_in[0];
  const int*   ei = (const int*)d_in[1];   // JAX default x64-off: int32 indices
  const float* Wv[4]  = {(const float*)d_in[2],  (const float*)d_in[6],
                         (const float*)d_in[10], (const float*)d_in[14]};
  const float* Asv[4] = {(const float*)d_in[3],  (const float*)d_in[7],
                         (const float*)d_in[11], (const float*)d_in[15]};
  const float* Adv[4] = {(const float*)d_in[4],  (const float*)d_in[8],
                         (const float*)d_in[12], (const float*)d_in[16]};
  const float* Bv[4]  = {(const float*)d_in[5],  (const float*)d_in[9],
                         (const float*)d_in[13], (const float*)d_in[17]};

  const int N    = in_sizes[0] / 128;
  const int E    = in_sizes[1] / 2;
  const int Etot = E + N;

  float* ws   = (float*)d_ws;
  float* hbuf = ws;                          // N*64 : GEMM output h (fp32)
  float* obuf = hbuf + (size_t)N * 64;       // N*64 : aggregation accumulator
  float* als  = obuf + (size_t)N * 64;       // N*8
  float* ald  = als + (size_t)N * 8;         // N*8
  float* mbuf = ald + (size_t)N * 8;         // N*8
  float* sbuf = mbuf + (size_t)N * 8;        // N*8
  unsigned short* Wt   = (unsigned short*)(sbuf + (size_t)N * 8);  // <= 128*64 bf16
  unsigned short* abuf = Wt + 128 * 64;      // N*128 bf16 activations (GEMM A)

  const int Kdim[4] = {128, 64, 64, 64};
  const int Hh[4]   = {8, 8, 8, 1};
  const int Dd[4]   = {8, 8, 8, 16};
  const int HDl[4]  = {64, 64, 64, 16};
  const int mTiles  = N / 16;                // 50000/16 = 3125 exactly

  // one-time: x (fp32) -> bf16 activation buffer
  f32_to_bf16<<<4096, 256, 0, stream>>>(x, (unsigned*)abuf, (N * 128) / 4);

  for (int L = 0; L < 4; ++L) {
    const int K = Kdim[L], H = Hh[L], D = Dd[L], hd = HDl[L];

    convert_wt<<<(K * hd + 255) / 256, 256, 0, stream>>>(Wv[L], Wt, K, hd);

    const int gblocks = (mTiles + 7) / 8;    // 8 waves (16 rows each) per WG
    if (L == 0)      gemm_wmma<128, 4><<<gblocks, 256, 0, stream>>>(abuf, Wt, hbuf, mTiles, hd);
    else if (L < 3)  gemm_wmma<64, 4><<<gblocks, 256, 0, stream>>>(abuf, Wt, hbuf, mTiles, hd);
    else             gemm_wmma<64, 1><<<gblocks, 256, 0, stream>>>(abuf, Wt, hbuf, mTiles, hd);

    const int nh = N * H;
    node_attn<<<(nh + 255) / 256, 256, 0, stream>>>(hbuf, Asv[L], Adv[L], als, ald, N, H, D);
    init_ms<<<(nh + 255) / 256, 256, 0, stream>>>(mbuf, sbuf, nh);

    edge_max<<<16384, 256, 0, stream>>>(ei, als, ald, mbuf, E, Etot, H);
    edge_sum<<<16384, 256, 0, stream>>>(ei, als, ald, mbuf, sbuf, E, Etot, H);

    float* outp = (L == 3) ? (float*)d_out : obuf;
    zero_buf<<<4096, 256, 0, stream>>>(outp, N * hd);
    edge_agg<<<32768, 256, 0, stream>>>(ei, hbuf, als, ald, mbuf, sbuf, outp, E, Etot, H, D);

    if (L < 3)
      bias_act_bf16<<<4096, 256, 0, stream>>>(obuf, Bv[L], abuf, N * hd, hd);
    else
      bias_final<<<4096, 256, 0, stream>>>((float*)d_out, Bv[L], N * hd, hd);
  }
}
